// MultiLevelGraphNetWithEdgeFeatures_62749472194934
// MI455X (gfx1250) — matrix operand order repaired and checked by reading, not compiled, not run on hardware
//
#include <hip/hip_runtime.h>

#define NN 4096   // nodes
#define NE 8192   // edges
#define NB 128    // graphs
#define KP 4224   // 32*128 + 128 bias rows for fused NNConv GEMM
#define NCH (KP / 32)

typedef __attribute__((ext_vector_type(16))) __bf16 v16bf;
typedef __attribute__((ext_vector_type(8)))  float  v8f;
union Frag { v16bf v; unsigned u[8]; };

__device__ __forceinline__ unsigned short f2bf(float f) {
    unsigned u = __float_as_uint(f);
    u += 0x7FFFu + ((u >> 16) & 1u);          // round-to-nearest-even
    return (unsigned short)(u >> 16);
}
__device__ __forceinline__ float bf2f(unsigned short h) {
    return __uint_as_float(((unsigned)h) << 16);
}
__device__ __forceinline__ float sigm(float x) { return 1.0f / (1.0f + expf(-x)); }

// ---------------------------------------------------------------------------
// Generic bf16 WMMA GEMM: C[M,N] = act(A[M,K] @ B[K,N] + bias)
// A: bf16 row-major (K multiple of 32, zero-padded). Bt: bf16, N x K (pre-T).
// Each wave computes one 16x16 tile. act: 0=none 1=relu 2=lecun_tanh 3=tanh
// ---------------------------------------------------------------------------
__global__ void gemm_bf16_kernel(const unsigned short* __restrict__ A,
                                 const unsigned short* __restrict__ Bt,
                                 const float* __restrict__ bias,
                                 float* __restrict__ C,
                                 unsigned short* __restrict__ Cb,
                                 int M, int N, int K, int act) {
    int tid = threadIdx.x, wave = tid >> 5, lane = tid & 31;
    int tilesN = (N + 15) >> 4, tilesM = (M + 15) >> 4;
    int tile = blockIdx.x * 4 + wave;
    if (tile >= tilesM * tilesN) return;
    int tm = tile / tilesN, tn = tile % tilesN;
    int m16 = lane & 15, h = lane >> 4;

    int rowA = tm * 16 + m16; if (rowA >= M) rowA = M - 1;
    int rowB = tn * 16 + m16; if (rowB >= N) rowB = N - 1;
    const unsigned short* Ap = A + (size_t)rowA * K;
    const unsigned short* Bp = Bt + (size_t)rowB * K;

    v8f acc = {0.f,0.f,0.f,0.f,0.f,0.f,0.f,0.f};
    for (int kc = 0; kc < K; kc += 32) {
        Frag a, b;
#pragma unroll
        for (int v = 0; v < 8; v++) {
            int ka = ((v >> 2) << 4) + (h << 3) + ((v & 3) << 1) + kc; // A frag layout
            a.u[v] = *(const unsigned*)(Ap + ka);
            int kb = ((v >> 2) << 4) + (h << 3) + ((v & 3) << 1) + kc; // B frag layout
            b.u[v] = *(const unsigned*)(Bp + kb);
        }
        acc = __builtin_amdgcn_wmma_f32_16x16x32_bf16(false, a.v, false, b.v,
                                                      (short)0, acc, false, false);
    }
    int nG = tn * 16 + m16;
#pragma unroll
    for (int r = 0; r < 8; r++) {
        int mG = tm * 16 + r + 8 * h;
        if (mG < M && nG < N) {
            float v = acc[r] + (bias ? bias[nG] : 0.0f);
            if (act == 1) v = fmaxf(v, 0.0f);
            else if (act == 2) v = 1.7159f * tanhf(0.666f * v);
            else if (act == 3) v = tanhf(v);
            C[(size_t)mG * N + nG] = v;
            if (Cb) Cb[(size_t)mG * N + nG] = f2bf(v);
        }
    }
}

// ---------------------------------------------------------------------------
// Fused NNConv message GEMM: msg[e,o] = sum_k A[e,k]*Bpt[o,k], A built on the
// fly in LDS: k<4096 -> h[e,k>>7]*x[e,k&127]; k>=4096 -> x[e,k-4096] (bias).
// B chunks are double-buffered into LDS with GLOBAL_LOAD_ASYNC_TO_LDS_B128
// (ASYNCcnt), overlapping the DMA with A-chunk construction and WMMA.
// Epilogue: atomic scatter-add into acc_nodes[dst[e]] (fused segment_sum).
// ---------------------------------------------------------------------------
__global__ void nnconv_msg_kernel(const unsigned short* __restrict__ Xbf,  // NN x 128
                                  const unsigned short* __restrict__ Hbf,  // NE x 32
                                  const unsigned short* __restrict__ Bpt,  // 128 x KP
                                  const int* __restrict__ src,
                                  const int* __restrict__ dst,
                                  float* __restrict__ acc_nodes) {
    __shared__ __attribute__((aligned(16))) unsigned short sX[64 * 128];
    __shared__ __attribute__((aligned(16))) unsigned short sH[64 * 32];
    __shared__ __attribute__((aligned(16))) unsigned short sA[64 * 32];
    __shared__ __attribute__((aligned(16))) unsigned short sB[2][128 * 32];
    __shared__ int sDst[64];
    int tid = threadIdx.x, wave = tid >> 5, lane = tid & 31;
    int eBase = blockIdx.x * 64;

    // Stage one 128x32 bf16 B-chunk into LDS: 512 x 16B lane-transfers,
    // 4 async instructions per wave (ASYNCcnt += 4 per wave).
    auto issueB = [&](int chunk, int buf) {
        int kcs = chunk * 32;
#pragma unroll
        for (int i = 0; i < 4; i++) {
            int T = tid + i * 128;            // 0..511
            int o = T >> 2, part = T & 3;
            const unsigned short* g = Bpt + (size_t)o * KP + kcs + part * 8;
            unsigned lds = (unsigned)(unsigned long long)(&sB[buf][o * 32 + part * 8]);
            asm volatile("global_load_async_to_lds_b128 %0, %1, off"
                         :: "v"(lds), "v"(g) : "memory");
        }
    };

    issueB(0, 0);   // prefetch first B chunk while we gather X/H rows

    for (int r = wave; r < 64; r += 4) {
        int e = eBase + r;
        int sv = src[e];
        const unsigned* xp = (const unsigned*)(Xbf + (size_t)sv * 128);
        unsigned* dx = (unsigned*)(sX + r * 128);
        dx[lane] = xp[lane];
        dx[lane + 32] = xp[lane + 32];
        if (lane < 16) ((unsigned*)(sH + r * 32))[lane] =
            ((const unsigned*)(Hbf + (size_t)e * 32))[lane];
        if (lane == 0) sDst[r] = dst[e];
    }
    __syncthreads();

    v8f zero8 = {0.f,0.f,0.f,0.f,0.f,0.f,0.f,0.f};
    v8f acc[8];
#pragma unroll
    for (int i = 0; i < 8; i++) acc[i] = zero8;

    int m16 = lane & 15, h = lane >> 4;
    int myRow = wave * 16 + m16;

    for (int c = 0; c < NCH; c++) {
        int cur = c & 1;
        if (c + 1 < NCH) issueB(c + 1, cur ^ 1);       // prefetch next chunk
        int kc = c * 32;
        for (int idx = tid; idx < 64 * 32; idx += 128) {  // build A = h (x) x
            int r = idx >> 5, kk = idx & 31, k = kc + kk;
            float val;
            if (k < 4096) val = bf2f(sH[r * 32 + (k >> 7)]) * bf2f(sX[r * 128 + (k & 127)]);
            else          val = bf2f(sX[r * 128 + (k - 4096)]);
            sA[idx] = f2bf(val);
        }
        if (c + 1 < NCH) asm volatile("s_wait_asynccnt 0x4" ::: "memory");
        else             asm volatile("s_wait_asynccnt 0x0" ::: "memory");
        __syncthreads();
        Frag a;
#pragma unroll
        for (int v = 0; v < 8; v++) {
            int ka = ((v >> 2) << 4) + (h << 3) + ((v & 3) << 1);
            a.u[v] = *(const unsigned*)(sA + myRow * 32 + ka);
        }
#pragma unroll
        for (int nt = 0; nt < 8; nt++) {
            Frag b;
            const unsigned short* bp = &sB[cur][(nt * 16 + m16) * 32];
#pragma unroll
            for (int v = 0; v < 8; v++)
                b.u[v] = *(const unsigned*)(bp + ((v >> 2) << 4) + (h << 3) + ((v & 3) << 1));
            acc[nt] = __builtin_amdgcn_wmma_f32_16x16x32_bf16(false, a.v, false, b.v,
                                                              (short)0, acc[nt], false, false);
        }
        __syncthreads();
    }
#pragma unroll
    for (int nt = 0; nt < 8; nt++)
#pragma unroll
        for (int r = 0; r < 8; r++) {
            int m = wave * 16 + r + 8 * h;
            int o = nt * 16 + m16;
            atomicAdd(acc_nodes + (size_t)sDst[m] * 128 + o, acc[nt][r]);
        }
}

// ---------------------------------------------------------------------------
// Front-end: x1 = relu(lin(x[:, :42])), 5-step CfC-1 recurrence, x2o.
// One block (128 threads) per node.
// ---------------------------------------------------------------------------
__global__ void frontend_kernel(const float* __restrict__ x,
                                const float* __restrict__ x11w, const float* __restrict__ x11b,
                                const float* __restrict__ bbw,  const float* __restrict__ bbb,
                                const float* __restrict__ f1w,  const float* __restrict__ f1b,
                                const float* __restrict__ f2w,  const float* __restrict__ f2b,
                                const float* __restrict__ taw,  const float* __restrict__ tab,
                                const float* __restrict__ tbw,  const float* __restrict__ tbb,
                                const float* __restrict__ x22w, const float* __restrict__ x22b,
                                float* __restrict__ x1o, float* __restrict__ x2oo) {
    __shared__ float sIn[48], sH[12], sCat[18], sZ[128], sG[48], sSteps[30];
    int node = blockIdx.x, t = threadIdx.x;
    if (t < 48) sIn[t] = x[node * 48 + t];
    __syncthreads();
    if (t < 32) {
        float a = x11b[t];
        for (int i = 0; i < 42; i++) a += sIn[i] * x11w[i * 32 + t];
        x1o[node * 32 + t] = fmaxf(a, 0.f);
    }
    if (t < 12) sH[t] = sIn[42 + (t % 6)];
    __syncthreads();
    for (int step = 0; step < 5; step++) {
        if (t < 18) sCat[t] = (t < 6) ? sIn[42 + t] : sH[t - 6];
        __syncthreads();
        if (t < 128) {
            float a = bbb[t];
            for (int i = 0; i < 18; i++) a += sCat[i] * bbw[i * 128 + t];
            sZ[t] = 1.7159f * tanhf(0.666f * a);
        }
        __syncthreads();
        if (t < 48) {
            int which = t / 12, j = t % 12;
            const float* w = which == 0 ? f1w : which == 1 ? f2w : which == 2 ? taw : tbw;
            const float* b = which == 0 ? f1b : which == 1 ? f2b : which == 2 ? tab : tbb;
            float a = b[j];
            for (int c = 0; c < 128; c++) a += sZ[c] * w[c * 12 + j];
            sG[t] = a;
        }
        __syncthreads();
        if (t < 12) {
            float ff1 = tanhf(sG[t]), ff2 = tanhf(sG[12 + t]);
            float tt = sigm(sG[24 + t] + sG[36 + t]);
            float hv = ff1 * (1.f - tt) + tt * ff2;
            sH[t] = hv;
            if (t < 6) sSteps[step * 6 + t] = hv;
        }
        __syncthreads();
    }
    if (t < 32) {
        float a = x22b[t];
        for (int i = 0; i < 30; i++) a += sSteps[i] * x22w[i * 32 + t];
        x2oo[node * 32 + t] = fmaxf(a, 0.f);
    }
}

// Attention over 2 tokens, 4 heads of dim 8. One thread per (node, head).
__global__ void mha2_kernel(const float* __restrict__ q, const float* __restrict__ k,
                            const float* __restrict__ v, unsigned short* __restrict__ outb) {
    int idx = blockIdx.x * 256 + threadIdx.x;
    if (idx >= NN * 4) return;
    int n = idx >> 2, hd = (idx & 3) * 8;
    int b0 = n * 32 + hd, b1 = (NN + n) * 32 + hd;
    float k0[8], k1[8], v0[8], v1[8];
    for (int d = 0; d < 8; d++) { k0[d]=k[b0+d]; k1[d]=k[b1+d]; v0[d]=v[b0+d]; v1[d]=v[b1+d]; }
    for (int s = 0; s < 2; s++) {
        int bq = s ? b1 : b0;
        float s0 = 0.f, s1 = 0.f;
        for (int d = 0; d < 8; d++) { float qd = q[bq + d]; s0 += qd * k0[d]; s1 += qd * k1[d]; }
        s0 *= 0.35355339059f; s1 *= 0.35355339059f;
        float mx = fmaxf(s0, s1);
        float e0 = expf(s0 - mx), e1 = expf(s1 - mx), inv = 1.f / (e0 + e1);
        e0 *= inv; e1 *= inv;
        for (int d = 0; d < 8; d++) outb[bq + d] = f2bf(e0 * v0[d] + e1 * v1[d]);
    }
}

// out = LayerNorm(x + res) * g + b ; dim 32 ; one thread per row
__global__ void ln_kernel(const float* __restrict__ x, const float* __restrict__ res,
                          const float* __restrict__ g, const float* __restrict__ b,
                          float* __restrict__ outf, unsigned short* __restrict__ outb, int rows) {
    int r = blockIdx.x * 256 + threadIdx.x;
    if (r >= rows) return;
    float buf[32], m = 0.f;
    for (int j = 0; j < 32; j++) { buf[j] = x[r * 32 + j] + res[r * 32 + j]; m += buf[j]; }
    m *= (1.f / 32.f);
    float var = 0.f;
    for (int j = 0; j < 32; j++) { float d = buf[j] - m; var += d * d; }
    var *= (1.f / 32.f);
    float inv = rsqrtf(var + 1e-5f);
    for (int j = 0; j < 32; j++) {
        float o = (buf[j] - m) * inv * g[j] + b[j];
        outf[r * 32 + j] = o;
        if (outb) outb[r * 32 + j] = f2bf(o);
    }
}

// -------------------- small elementwise / packing kernels --------------------
__global__ void build_mem_kernel(const float* x1, const float* x2o,
                                 float* mem, unsigned short* memb) {
    int i = blockIdx.x * 256 + threadIdx.x; if (i >= NN * 32) return;
    float a = x1[i], b = x2o[i];
    mem[i] = a; mem[i + NN * 32] = b; memb[i] = f2bf(a); memb[i + NN * 32] = f2bf(b);
}
__global__ void tgt_init_kernel(const float* mem, float* tgt, unsigned short* tgtb) {
    int i = blockIdx.x * 256 + threadIdx.x; if (i >= NN * 32) return;
    float v = 0.5f * (mem[i] + mem[i + NN * 32]);
    tgt[i] = v; tgt[i + NN * 32] = v;
    unsigned short h = f2bf(v); tgtb[i] = h; tgtb[i + NN * 32] = h;
}
__global__ void tout_kernel(const float* tgt, float* tout) {
    int i = blockIdx.x * 256 + threadIdx.x; if (i >= NN * 32) return;
    tout[i] = fmaxf(0.f, 0.5f * (tgt[i] + tgt[i + NN * 32]));
}
__global__ void build_bbin_kernel(const float* x1, const float* x2o, const float* tout,
                                  unsigned short* bbin) {
    int i = blockIdx.x * 256 + threadIdx.x; if (i >= NN * 128) return;
    int n = i >> 7, c = i & 127;
    float v = (c < 32) ? x1[n * 32 + c] : (c < 64) ? x2o[n * 32 + c - 32]
            : (c < 96) ? tout[n * 32 + c - 64] : tout[n * 32 + c - 96];
    bbin[i] = f2bf(v);
}
__global__ void cfc2_gate_kernel(const float* cf1, const float* cf2, const float* cta,
                                 const float* ctb, float* out) {
    int i = blockIdx.x * 256 + threadIdx.x; if (i >= NN * 64) return;
    float t = sigm(cta[i] + ctb[i]);
    out[i] = fmaxf(0.f, cf1[i] * (1.f - t) + t * cf2[i]);
}
__global__ void build_xm_kernel(const float* cfco, const float* x1, const float* x2o,
                                unsigned short* xmb) {
    int i = blockIdx.x * 256 + threadIdx.x; if (i >= NN * 128) return;
    int n = i >> 7, c = i & 127;
    float v = (c < 64) ? cfco[n * 64 + c] : (c < 96) ? x1[n * 32 + c - 64] : x2o[n * 32 + c - 96];
    xmb[i] = f2bf(v);
}
__global__ void cnt_kernel(const int* dst, float* cnt) {
    int e = blockIdx.x * 256 + threadIdx.x; if (e >= NE) return;
    atomicAdd(cnt + dst[e], 1.0f);
}
__global__ void conv_combine_kernel(const float* acc, const float* cnt, const float* rooto,
                                    const float* bias, float* sxf, unsigned short* sxb, int relu) {
    int i = blockIdx.x * 256 + threadIdx.x; if (i >= NN * 128) return;
    float v = acc[i] / fmaxf(cnt[i >> 7], 1.0f) + rooto[i] + bias[i & 127];
    if (relu) v = fmaxf(v, 0.f);
    sxf[i] = v;
    if (sxb) sxb[i] = f2bf(v);
}
__global__ void zero_f32_kernel(float* p, int n) {
    int i = blockIdx.x * 256 + threadIdx.x; if (i < n) p[i] = 0.f;
}
__global__ void fill_u32_kernel(unsigned* p, unsigned v, int n) {
    int i = blockIdx.x * 256 + threadIdx.x; if (i < n) p[i] = v;
}
__global__ void convert_pad_kernel(const float* src, unsigned short* dst, int rows, int K, int Kp) {
    int i = blockIdx.x * 256 + threadIdx.x; if (i >= rows * Kp) return;
    int r = i / Kp, kp = i - r * Kp;
    dst[i] = (kp < K) ? f2bf(src[r * K + kp]) : (unsigned short)0;
}
__global__ void transpose_w_kernel(const float* W, unsigned short* Wt, int K, int N, int Kp) {
    int i = blockIdx.x * 256 + threadIdx.x; if (i >= N * Kp) return;
    int n = i / Kp, kp = i - n * Kp;
    Wt[i] = (kp < K) ? f2bf(W[kp * N + n]) : (unsigned short)0;
}
// Repack e2 weights: Bpt[o][k] : k<4096 -> W2[k>>7][(k&127)*128+o] ; else b2[(k-4096)*128+o]
__global__ void bpt_kernel(const float* W2, const float* b2, unsigned short* Bpt) {
    int i = blockIdx.x * 256 + threadIdx.x; if (i >= 128 * KP) return;
    int o = i / KP, k = i - o * KP;
    float v = (k < 4096) ? W2[(k >> 7) * 16384 + (k & 127) * 128 + o] : b2[(k - 4096) * 128 + o];
    Bpt[i] = f2bf(v);
}

// ---------------------------- Set2Set kernels ------------------------------
__global__ void s2s_g_kernel(const float* qs, const float* hh, const float* wih,
                             const float* whh, const float* bih, const float* bhh, float* g) {
    int i = blockIdx.x * 256 + threadIdx.x; if (i >= NB * 512) return;
    int b = i >> 9, j = i & 511;
    float a = bih[j] + bhh[j];
    for (int k = 0; k < 256; k++) a += qs[b * 256 + k] * wih[k * 512 + j];
    for (int k = 0; k < 128; k++) a += hh[b * 128 + k] * whh[k * 512 + j];
    g[i] = a;
}
__global__ void s2s_lstm_kernel(const float* g, float* cc, float* hh) {
    int i = blockIdx.x * 256 + threadIdx.x; if (i >= NB * 128) return;
    int b = i >> 7, u = i & 127;
    float gi = g[b * 512 + u], gf = g[b * 512 + 128 + u];
    float gg = g[b * 512 + 256 + u], go = g[b * 512 + 384 + u];
    float c = sigm(gf) * cc[i] + sigm(gi) * tanhf(gg);
    cc[i] = c;
    hh[i] = sigm(go) * tanhf(c);
}
__device__ __forceinline__ unsigned fenc(float f) {
    unsigned u = __float_as_uint(f);
    return (u & 0x80000000u) ? ~u : (u | 0x80000000u);
}
__device__ __forceinline__ float fdec(unsigned u) {
    return __uint_as_float((u & 0x80000000u) ? (u & 0x7FFFFFFFu) : ~u);
}
__global__ void s2s_e_kernel(const float* sx, const float* hh, const int* batch,
                             float* ebuf, unsigned* emax) {
    int v = blockIdx.x * 256 + threadIdx.x; if (v >= NN) return;
    int b = batch[v];
    float e = 0.f;
    for (int k = 0; k < 128; k++) e += sx[v * 128 + k] * hh[b * 128 + k];
    ebuf[v] = e;
    atomicMax(emax + b, fenc(e));
}
__global__ void s2s_a_kernel(const float* ebuf, const unsigned* emax, const int* batch,
                             float* abuf, float* asum) {
    int v = blockIdx.x * 256 + threadIdx.x; if (v >= NN) return;
    int b = batch[v];
    float a = expf(ebuf[v] - fdec(emax[b]));
    abuf[v] = a;
    atomicAdd(asum + b, a);
}
__global__ void s2s_r_kernel(const float* abuf, const float* asum, const float* sx,
                             const int* batch, float* r) {
    int i = blockIdx.x * 256 + threadIdx.x; if (i >= NN * 128) return;
    int v = i >> 7, o = i & 127, b = batch[v];
    atomicAdd(r + b * 128 + o, (abuf[v] / asum[b]) * sx[i]);
}
__global__ void s2s_q_kernel(const float* hh, const float* r, float* qs) {
    int i = blockIdx.x * 256 + threadIdx.x; if (i >= NB * 128) return;
    int b = i >> 7, u = i & 127;
    qs[b * 256 + u] = hh[i];
    qs[b * 256 + 128 + u] = r[i];
}

// ===========================================================================
extern "C" void kernel_launch(void* const* d_in, const int* in_sizes, int n_in,
                              void* d_out, int out_size, void* d_ws, size_t ws_size,
                              hipStream_t stream) {
    (void)in_sizes; (void)n_in; (void)out_size; (void)ws_size;
    auto F = [&](int i) { return (const float*)d_in[i]; };
    const float* x     = F(0);
    const int*   eidx  = (const int*)d_in[1];
    const float* eattr = F(2);
    const int*   batch = (const int*)d_in[3];
    const int* src = eidx;
    const int* dst = eidx + NE;

    // ---- param index map (jax tree_flatten: sorted dict keys, leaves b<w) ----
    // 4..13 cfc1{bb,ff1,ff2,ta,tb}(b,w) ; 14..23 cfc2 ; 24..29 conv1{bias,e1.b,
    // e1.w,e2.b,e2.w,root} ; 30..35 conv2 ; 36..87 dec[2]{ca(k,o,q,v)(b,w),ff1,
    // ff2,ln1(b,g),ln2,ln3,sa} ; 88..119 enc[2]{attn,ff1,ff2,ln1,ln2} ;
    // 120..127 fc1..fc4(b,w) ; 128..131 s2s{bhh,bih,whh,wih} ; 132..135 x11,x22
    char* wp = (char*)d_ws;
    auto alloc  = [&](size_t b) { void* p = (void*)wp; wp += (b + 255) & ~(size_t)255; return p; };
    auto allocF = [&](size_t n) { return (float*)alloc(n * 4); };
    auto allocH = [&](size_t n) { return (unsigned short*)alloc(n * 2); };

    auto TW = [&](int wi, int K, int N, int Kp) {
        unsigned short* p = allocH((size_t)N * Kp);
        int n = N * Kp;
        transpose_w_kernel<<<(n + 255) / 256, 256, 0, stream>>>(F(wi), p, K, N, Kp);
        return p;
    };
    auto GEMM = [&](const unsigned short* A, const unsigned short* Bt, const float* bias,
                    float* C, unsigned short* Cb, int M, int N, int K, int act) {
        int tiles = ((M + 15) / 16) * ((N + 15) / 16);
        gemm_bf16_kernel<<<(tiles + 3) / 4, 128, 0, stream>>>(A, Bt, bias, C, Cb, M, N, K, act);
    };
    auto ZERO = [&](float* p, int n) { zero_f32_kernel<<<(n + 255) / 256, 256, 0, stream>>>(p, n); };

    // ------------------------- weight preprocessing -------------------------
    struct MhaP { const float *kb, *ob, *qb, *vb; unsigned short *kt, *ot, *qt, *vt; };
    auto mkMha = [&](int base) {
        MhaP m;
        m.kb = F(base + 0); m.kt = TW(base + 1, 32, 32, 32);
        m.ob = F(base + 2); m.ot = TW(base + 3, 32, 32, 32);
        m.qb = F(base + 4); m.qt = TW(base + 5, 32, 32, 32);
        m.vb = F(base + 6); m.vt = TW(base + 7, 32, 32, 32);
        return m;
    };
    MhaP encA[2], decSA[2], decCA[2];
    unsigned short *encF1t[2], *encF2t[2], *decF1t[2], *decF2t[2];
    for (int L = 0; L < 2; L++) {
        int eb = 88 + 16 * L, db = 36 + 26 * L;
        encA[L]  = mkMha(eb);
        encF1t[L] = TW(eb + 9, 32, 64, 32);
        encF2t[L] = TW(eb + 11, 64, 32, 64);
        decCA[L] = mkMha(db);
        decSA[L] = mkMha(db + 18);
        decF1t[L] = TW(db + 9, 32, 64, 32);
        decF2t[L] = TW(db + 11, 64, 32, 64);
    }
    unsigned short* bbT  = TW(15, 128, 128, 128);
    unsigned short* cf1T = TW(17, 128, 64, 128);
    unsigned short* cf2T = TW(19, 128, 64, 128);
    unsigned short* ctaT = TW(21, 128, 64, 128);
    unsigned short* ctbT = TW(23, 128, 64, 128);
    unsigned short* e1T1 = TW(26, 10, 32, 32);
    unsigned short* e1T2 = TW(32, 10, 32, 32);
    unsigned short* rootT1 = TW(29, 128, 128, 128);
    unsigned short* rootT2 = TW(35, 128, 128, 128);
    unsigned short* fc1T = TW(121, 256, 512, 256);
    unsigned short* fc2T = TW(123, 512, 128, 512);
    unsigned short* fc3T = TW(125, 128, 64, 128);
    unsigned short* fc4T = TW(127, 64, 1, 64);
    unsigned short* Bpt1 = allocH((size_t)128 * KP);
    unsigned short* Bpt2 = allocH((size_t)128 * KP);
    bpt_kernel<<<(128 * KP + 255) / 256, 256, 0, stream>>>(F(28), F(27), Bpt1);
    bpt_kernel<<<(128 * KP + 255) / 256, 256, 0, stream>>>(F(34), F(33), Bpt2);

    // ------------------------------- buffers --------------------------------
    float *x1f = allocF(NN * 32), *x2of = allocF(NN * 32);
    float *mem = allocF(2 * NN * 32);  unsigned short *memb = allocH(2 * NN * 32);
    float *tgt = allocF(2 * NN * 32);  unsigned short *tgtb = allocH(2 * NN * 32);
    float *qb = allocF(2 * NN * 32), *kb = allocF(2 * NN * 32), *vb = allocF(2 * NN * 32);
    unsigned short *aob = allocH(2 * NN * 32);
    float *proj = allocF(2 * NN * 64); unsigned short *projb = allocH(2 * NN * 64);
    float *toutf = allocF(NN * 32);
    unsigned short *bbin = allocH(NN * 128);
    float *zf = allocF(NN * 128);      unsigned short *zbf = allocH(NN * 128);
    float *cf1 = allocF(NN * 64), *cf2 = allocF(NN * 64);
    float *cta = allocF(NN * 64), *ctb = allocF(NN * 64);
    float *cfco = allocF(NN * 64);
    unsigned short *xmb = allocH(NN * 128);
    unsigned short *eab = allocH(NE * 32);
    float *h1f = allocF(NE * 32);      unsigned short *h1b = allocH(NE * 32);
    float *h2f = allocF(NE * 32);      unsigned short *h2b = allocH(NE * 32);
    float *accN = allocF(NN * 128), *rooto = allocF(NN * 128), *cntN = allocF(NN);
    float *sxA = allocF(NN * 128);     unsigned short *sxAb = allocH(NN * 128);
    float *sxB = allocF(NN * 128);     unsigned short *sxBb = allocH(NN * 128);
    float *sxF = allocF(NN * 128);
    float *qs = allocF(NB * 256), *hh = allocF(NB * 128), *cc = allocF(NB * 128);
    float *gbuf = allocF(NB * 512), *ebuf = allocF(NN), *abuf = allocF(NN);
    float *asum = allocF(NB), *rbuf = allocF(NB * 128);
    unsigned *emax = (unsigned*)alloc(NB * 4);
    unsigned short *qsb = allocH(NB * 256);
    float *y1 = allocF(NB * 512);      unsigned short *y1b = allocH(NB * 512);
    float *y2 = allocF(NB * 128);      unsigned short *y2b = allocH(NB * 128);
    float *y3 = allocF(NB * 64);       unsigned short *y3b = allocH(NB * 64);

    // ------------------------------ front-end -------------------------------
    frontend_kernel<<<NN, 128, 0, stream>>>(x, F(133), F(132), F(5), F(4), F(7), F(6),
                                            F(9), F(8), F(11), F(10), F(13), F(12),
                                            F(135), F(134), x1f, x2of);
    build_mem_kernel<<<(NN * 32 + 255) / 256, 256, 0, stream>>>(x1f, x2of, mem, memb);

    // ------------------------- encoder / decoder ----------------------------
    auto runMha = [&](const MhaP& m, unsigned short* qin, unsigned short* kvin,
                      float* resf, unsigned short* resb, const float* lnb, const float* lng) {
        GEMM(qin, m.qt, m.qb, qb, nullptr, 2 * NN, 32, 32, 0);
        GEMM(kvin, m.kt, m.kb, kb, nullptr, 2 * NN, 32, 32, 0);
        GEMM(kvin, m.vt, m.vb, vb, nullptr, 2 * NN, 32, 32, 0);
        mha2_kernel<<<(NN * 4 + 255) / 256, 256, 0, stream>>>(qb, kb, vb, aob);
        GEMM(aob, m.ot, m.ob, proj, nullptr, 2 * NN, 32, 32, 0);
        ln_kernel<<<(2 * NN + 255) / 256, 256, 0, stream>>>(resf, proj, lng, lnb, resf, resb, 2 * NN);
    };
    auto runFF = [&](unsigned short* f1t, unsigned short* f2t, const float* b1, const float* b2,
                     float* resf, unsigned short* resb, const float* lnb, const float* lng) {
        GEMM(resb, f1t, b1, proj, projb, 2 * NN, 64, 32, 1);
        GEMM(projb, f2t, b2, qb, nullptr, 2 * NN, 32, 64, 0);
        ln_kernel<<<(2 * NN + 255) / 256, 256, 0, stream>>>(resf, qb, lng, lnb, resf, resb, 2 * NN);
    };
    for (int L = 0; L < 2; L++) {
        int eb = 88 + 16 * L;
        runMha(encA[L], memb, memb, mem, memb, F(eb + 12), F(eb + 13));
        runFF(encF1t[L], encF2t[L], F(eb + 8), F(eb + 10), mem, memb, F(eb + 14), F(eb + 15));
    }
    tgt_init_kernel<<<(NN * 32 + 255) / 256, 256, 0, stream>>>(mem, tgt, tgtb);
    for (int L = 0; L < 2; L++) {
        int db = 36 + 26 * L;
        runMha(decSA[L], tgtb, tgtb, tgt, tgtb, F(db + 12), F(db + 13));
        runMha(decCA[L], tgtb, memb, tgt, tgtb, F(db + 14), F(db + 15));
        runFF(decF1t[L], decF2t[L], F(db + 8), F(db + 10), tgt, tgtb, F(db + 16), F(db + 17));
    }
    tout_kernel<<<(NN * 32 + 255) / 256, 256, 0, stream>>>(tgt, toutf);

    // -------------------------------- CfC-2 ---------------------------------
    build_bbin_kernel<<<(NN * 128 + 255) / 256, 256, 0, stream>>>(x1f, x2of, toutf, bbin);
    GEMM(bbin, bbT, F(14), zf, zbf, NN, 128, 128, 2);          // z = lecun_tanh
    GEMM(zbf, cf1T, F(16), cf1, nullptr, NN, 64, 128, 3);      // tanh
    GEMM(zbf, cf2T, F(18), cf2, nullptr, NN, 64, 128, 3);      // tanh
    GEMM(zbf, ctaT, F(20), cta, nullptr, NN, 64, 128, 0);
    GEMM(zbf, ctbT, F(22), ctb, nullptr, NN, 64, 128, 0);
    cfc2_gate_kernel<<<(NN * 64 + 255) / 256, 256, 0, stream>>>(cf1, cf2, cta, ctb, cfco);
    build_xm_kernel<<<(NN * 128 + 255) / 256, 256, 0, stream>>>(cfco, x1f, x2of, xmb);

    // -------------------------------- NNConv --------------------------------
    convert_pad_kernel<<<(NE * 32 + 255) / 256, 256, 0, stream>>>(eattr, eab, NE, 10, 32);
    GEMM(eab, e1T1, F(25), h1f, h1b, NE, 32, 32, 1);
    GEMM(eab, e1T2, F(31), h2f, h2b, NE, 32, 32, 1);
    ZERO(cntN, NN);
    cnt_kernel<<<(NE + 255) / 256, 256, 0, stream>>>(dst, cntN);
    auto CONV = [&](unsigned short* xbf, unsigned short* hbf, unsigned short* bpt,
                    unsigned short* roott, const float* bias, float* sxf,
                    unsigned short* sxb, int relu) {
        ZERO(accN, NN * 128);
        nnconv_msg_kernel<<<NE / 64, 128, 0, stream>>>(xbf, hbf, bpt, src, dst, accN);
        GEMM(xbf, roott, nullptr, rooto, nullptr, NN, 128, 128, 0);
        conv_combine_kernel<<<(NN * 128 + 255) / 256, 256, 0, stream>>>(accN, cntN, rooto,
                                                                        bias, sxf, sxb, relu);
    };
    CONV(xmb,  h1b, Bpt1, rootT1, F(24), sxA, sxAb, 1);
    CONV(sxAb, h2b, Bpt2, rootT2, F(30), sxB, sxBb, 1);
    CONV(sxBb, h2b, Bpt2, rootT2, F(30), sxF, nullptr, 0);

    // -------------------------------- Set2Set -------------------------------
    ZERO(qs, NB * 256); ZERO(hh, NB * 128); ZERO(cc, NB * 128);
    for (int it = 0; it < 2; it++) {
        s2s_g_kernel<<<(NB * 512 + 255) / 256, 256, 0, stream>>>(qs, hh, F(131), F(130),
                                                                 F(129), F(128), gbuf);
        s2s_lstm_kernel<<<(NB * 128 + 255) / 256, 256, 0, stream>>>(gbuf, cc, hh);
        fill_u32_kernel<<<1, 256, 0, stream>>>(emax, 0x007FFFFFu, NB);  // enc(-inf)
        s2s_e_kernel<<<(NN + 255) / 256, 256, 0, stream>>>(sxF, hh, batch, ebuf, emax);
        ZERO(asum, NB);
        s2s_a_kernel<<<(NN + 255) / 256, 256, 0, stream>>>(ebuf, emax, batch, abuf, asum);
        ZERO(rbuf, NB * 128);
        s2s_r_kernel<<<(NN * 128 + 255) / 256, 256, 0, stream>>>(abuf, asum, sxF, batch, rbuf);
        s2s_q_kernel<<<(NB * 128 + 255) / 256, 256, 0, stream>>>(hh, rbuf, qs);
    }
    convert_pad_kernel<<<(NB * 256 + 255) / 256, 256, 0, stream>>>(qs, qsb, NB, 256, 256);

    // -------------------------------- MLP head ------------------------------
    GEMM(qsb, fc1T, F(120), y1, y1b, NB, 512, 256, 1);
    GEMM(y1b, fc2T, F(122), y2, y2b, NB, 128, 512, 1);
    GEMM(y2b, fc3T, F(124), y3, y3b, NB, 64, 128, 1);
    GEMM(y3b, fc4T, F(126), (float*)d_out, nullptr, NB, 1, 64, 0);
}
